// TreeLSTM_39616778338740
// MI455X (gfx1250) — compile-verified, matrix operand order checked by reading, not run on hardware
//
#include <hip/hip_runtime.h>

#define Ntot 2048
#define Kc   3
#define Hd   128
#define Vops 32
#define NT   4                 // node-tiles (16 rows each) per super-tile
#define APITCH 392             // LDS pitch (halves) for activation tiles (16B aligned, conflict-free)
#define WPITCH 136             // LDS pitch (halves) for weight tiles (272B rows)
#define USE_TDM 1              // 1: TENSOR_LOAD_TO_LDS weight staging (if builtin exists)

#if defined(__has_builtin)
#if __has_builtin(__builtin_amdgcn_tensor_load_to_lds) && __has_builtin(__builtin_amdgcn_s_wait_tensorcnt)
#define HAVE_TDM 1
#endif
#endif
#ifndef HAVE_TDM
#define HAVE_TDM 0
#endif

typedef __attribute__((ext_vector_type(16))) __bf16       v16bf;
typedef __attribute__((ext_vector_type(8)))  float        v8f;
typedef __attribute__((ext_vector_type(4)))  float        f32x4;
typedef __attribute__((ext_vector_type(4)))  unsigned int u32x4;
typedef __attribute__((ext_vector_type(2)))  unsigned int u32x2;
typedef __attribute__((ext_vector_type(8)))  int          i32x8;
typedef __attribute__((ext_vector_type(4)))  int          i32x4;

// 16-element bf16 WMMA fragment from LDS: elements 0..7 at p, 8..15 at p+16
// halves (16-bit A/B layout: a lane's second 8 K-values sit +16 in K).
__device__ __forceinline__ v16bf ld_frag(const __bf16* p) {
    union { v16bf v; u32x4 q[2]; } u;
    u.q[0] = *(const u32x4*)(p);
    u.q[1] = *(const u32x4*)(p + 16);
    return u.v;
}

__device__ __forceinline__ float sigmoidf_(float x) {
    return 1.0f / (1.0f + __expf(-x));
}

// ---- Weight staging: issue (no wait). dst is an LDS byte offset. ----------
__device__ __forceinline__ void stage_issue(const __bf16* __restrict__ wsrc,
                                            unsigned ldsDst, int tid) {
#if HAVE_TDM && USE_TDM
    if (tid < 32) {   // one wave issues one TDM op for the whole 128x128 tile
        unsigned long long ga = (unsigned long long)(size_t)wsrc;
        u32x4 g0;
        g0.x = 1u;                                        // count=1 (valid user D#)
        g0.y = ldsDst;                                    // lds_addr (bytes)
        g0.z = (unsigned)(ga & 0xFFFFFFFFu);              // global_addr[31:0]
        g0.w = (unsigned)((ga >> 32) & 0x01FFFFFFu)       // global_addr[56:32]
             | (2u << 30);                                // type=2 ("image")
        i32x8 g1;
        // data_size=1(2B) | pad_enable | pad_interval=5 (64 dwords) | pad_amount=3 (4 dwords)
        g1[0] = (int)((1u << 16) | (1u << 20) | (5u << 22) | (3u << 25));
        g1[1] = (int)(128u << 16);                        // tensor_dim0 = 128
        g1[2] = (int)(128u << 16);                        // tensor_dim1 = 128
        g1[3] = (int)(128u << 16);                        // tile_dim0   = 128
        g1[4] = 128;                                      // tile_dim1   = 128
        g1[5] = 128;                                      // tensor_dim0_stride = 128
        g1[6] = (int)((16384u & 0xFFFFu) << 16);          // tensor_dim1_stride lo
        g1[7] = 0;                                        // tensor_dim1_stride hi
        i32x4 gz  = {0, 0, 0, 0};
        i32x8 gz8 = {0, 0, 0, 0, 0, 0, 0, 0};
        // clang-23 lane: 6-arg form (u32x4, i32x8, i32x4, i32x4, i32x8, i32 cpol)
        __builtin_amdgcn_tensor_load_to_lds(g0, g1, gz, gz, gz8, 0);
    }
#else
    for (int c = tid; c < (Hd * Hd) / 8; c += 256) {
        int colW = c >> 4, chunk = c & 15;
        unsigned lo = ldsDst + (unsigned)(colW * (WPITCH * 2) + chunk * 16);
        asm volatile("global_load_async_to_lds_b128 %0, %1, off"
                     :: "v"(lo), "v"(wsrc + c * 8) : "memory");
    }
#endif
}

__device__ __forceinline__ void stage_wait(int tid) {
#if HAVE_TDM && USE_TDM
    if (tid < 32) __builtin_amdgcn_s_wait_tensorcnt(0);
#else
    (void)tid;
    asm volatile("s_wait_asynccnt 0x0" ::: "memory");
#endif
}

// ---------------------------------------------------------------------------
// Pre-pass: convert all W_{i,o,u,f} (V,K,H,H) f32 -> bf16 transposed into
// d_ws as Wt[g][v*K+k][col][row], g order {i,o,u,f}. 12.6 MB total.
// ---------------------------------------------------------------------------
__global__ __launch_bounds__(256) void prep_weights(
    const float* __restrict__ W_i, const float* __restrict__ W_o,
    const float* __restrict__ W_u, const float* __restrict__ W_f,
    __bf16* __restrict__ Wt)
{
    __shared__ __bf16 tile[Hd][WPITCH];
    const int mat = blockIdx.x;                  // g*(V*K) + (v*K + k)
    const int g   = mat / (Vops * Kc);
    const int vk  = mat % (Vops * Kc);
    const float* srcs[4] = { W_i, W_o, W_u, W_f };
    const float* src = srcs[g] + (size_t)vk * Hd * Hd;
    const int tid = threadIdx.x;

    for (int e4 = tid; e4 < (Hd * Hd) / 4; e4 += 256) {
        f32x4 wv = *(const f32x4*)(src + e4 * 4);      // row r, cols c..c+3
        int r = (e4 * 4) >> 7, c = (e4 * 4) & 127;
        tile[c + 0][r] = (__bf16)wv.x;
        tile[c + 1][r] = (__bf16)wv.y;
        tile[c + 2][r] = (__bf16)wv.z;
        tile[c + 3][r] = (__bf16)wv.w;
    }
    __syncthreads();
    __bf16* dst = Wt + (size_t)mat * Hd * Hd;          // [col][row] row-major
    for (int ch = tid; ch < (Hd * Hd) / 8; ch += 256) {
        int c = ch >> 4, o = (ch & 15) * 8;
        *(u32x4*)(dst + ch * 8) = *(const u32x4*)&tile[c][o];
    }
}

__device__ __forceinline__ size_t mat_off(int idx, int v) {
    int g = idx / 3, k = idx % 3;                      // g: 0=i 1=o 2=u 3=f
    return ((size_t)(g * Vops + v) * Kc + k) * (size_t)(Hd * Hd);
}

// ---------------------------------------------------------------------------
// Main kernel: one block per op value; 8 wave32s; each wave owns one 16-wide
// N tile of H=128. NT=4 node-tiles per super-tile; 12 weight matrices flow
// through a double-buffered LDS pipeline (TDM load overlapped with WMMA).
// ---------------------------------------------------------------------------
__global__ __launch_bounds__(256) void tree_lstm_wmma(
    const float* __restrict__ child_h, const float* __restrict__ child_c,
    const __bf16* __restrict__ Wt,
    const float* __restrict__ b_i, const float* __restrict__ b_f,
    const float* __restrict__ b_o, const float* __restrict__ b_u,
    const int* __restrict__ ops, const int* __restrict__ nch,
    float* __restrict__ out)
{
    __shared__ unsigned short s_idx[Ntot];
    __shared__ int            s_count;
    __shared__ __bf16         sAm[NT * 16][APITCH];    // masked h
    __shared__ __bf16         sAf[NT * 16][APITCH];    // unmasked h
    __shared__ __bf16         sW[2][Hd][WPITCH];       // double-buffered W^T tile
    __shared__ float          sMask[NT * 16][Kc];

    const int v    = blockIdx.x;
    const int tid  = threadIdx.x;
    const int lane = tid & 31;
    const int wave = tid >> 5;

    if (tid == 0) s_count = 0;
    __syncthreads();
    for (int n = tid; n < Ntot; n += 256)
        if (ops[n] == v) s_idx[atomicAdd(&s_count, 1)] = (unsigned short)n;
    __syncthreads();
    const int count = s_count;
    const int nsup  = (count + NT * 16 - 1) / (NT * 16);

    const int row = lane & 15;            // A-fragment row (M)
    const int kb  = (lane >> 4) << 3;     // K sub-offset per lane half
    const int col = (wave << 4) + (lane & 15);
    const int mhi = (lane >> 4) << 3;     // D layout: lanes 16-31 hold M = r+8

    const unsigned sWb0 = (unsigned)(size_t)(&sW[0][0][0]);
    const unsigned sWb1 = (unsigned)(size_t)(&sW[1][0][0]);

    // Per-lane biases (col fixed for the whole kernel).
    float bi_l[Kc], bo_l[Kc], bu_l[Kc], bf_l[Kc];
#pragma unroll
    for (int k = 0; k < Kc; ++k) {
        bi_l[k] = b_i[(v * Kc + k) * Hd + col];
        bo_l[k] = b_o[(v * Kc + k) * Hd + col];
        bu_l[k] = b_u[(v * Kc + k) * Hd + col];
        bf_l[k] = b_f[(v * Kc + k) * Hd + col];
    }

    for (int st = 0; st < nsup; ++st) {
        const int base = st * NT * 16;
        __syncthreads();   // previous super-tile fully consumed

        // ---- Stage A: bf16 activation tiles (masked + unmasked) ----
        for (int e4 = tid; e4 < (NT * 16 * Kc * Hd) / 4; e4 += 256) {
            int e  = e4 << 2;
            int m  = e / (Kc * Hd);
            int kk = e % (Kc * Hd);
            int k  = kk >> 7;
            int gm = base + m;
            f32x4 h4 = {0.f, 0.f, 0.f, 0.f};
            float mk = 0.f;
            if (gm < count) {
                int node = s_idx[gm];
                h4 = *(const f32x4*)(child_h + node * (Kc * Hd) + kk);
                mk = (k <= nch[node]) ? 1.f : 0.f;
            }
            union { __bf16 h[4]; u32x2 u; } pf, pm;
#pragma unroll
            for (int j = 0; j < 4; ++j) {
                pf.h[j] = (__bf16)h4[j];
                pm.h[j] = (__bf16)(h4[j] * mk);
            }
            *(u32x2*)&sAf[m][kk] = pf.u;
            *(u32x2*)&sAm[m][kk] = pm.u;
        }
        if (tid < NT * 16 * Kc) {
            int m = tid / Kc, k = tid % Kc;
            int gm = base + m;
            float mk = 0.f;
            if (gm < count) mk = (k <= nch[s_idx[gm]]) ? 1.f : 0.f;
            sMask[m][k] = mk;
        }
        __syncthreads();

        v8f accG[3][NT];   // i, o, u gate pre-activations per node-tile
        v8f accFC[NT];     // sum_k mask*f*c
#pragma unroll
        for (int t = 0; t < NT; ++t) {
            accG[0][t] = (v8f){};
            accG[1][t] = (v8f){};
            accG[2][t] = (v8f){};
            accFC[t]   = (v8f){};
        }

        // ---- Double-buffered weight pipeline over 12 (gate,child) matrices
        stage_issue(Wt + mat_off(0, v), sWb0, tid);
        stage_wait(tid);
        __syncthreads();
#pragma unroll
        for (int idx = 0; idx < 12; ++idx) {
            const int g   = idx / 3;           // 0=i 1=o 2=u 3=f
            const int k   = idx % 3;
            const int buf = idx & 1;
            if (idx < 11)   // prefetch next matrix into the other buffer
                stage_issue(Wt + mat_off(idx + 1, v),
                            (buf == 0) ? sWb1 : sWb0, tid);

            const __bf16 (*sWc)[WPITCH] = sW[buf];
            if (g < 3) {
#pragma unroll
                for (int t = 0; t < NT; ++t) {
                    v8f acc = accG[g][t];
#pragma unroll
                    for (int kkk = 0; kkk < 4; ++kkk) {
                        v16bf a = ld_frag(&sAm[t * 16 + row][k * 128 + kkk * 32 + kb]);
                        v16bf b = ld_frag(&sWc[col][kkk * 32 + kb]);
                        acc = __builtin_amdgcn_wmma_f32_16x16x32_bf16(
                            false, a, false, b, (short)0, acc, false, false);
                    }
                    accG[g][t] = acc;
                }
            } else {       // forget gate: per-child, immediate fc epilogue
#pragma unroll
                for (int t = 0; t < NT; ++t) {
                    v8f d = (v8f){};
#pragma unroll
                    for (int kkk = 0; kkk < 4; ++kkk) {
                        v16bf a = ld_frag(&sAf[t * 16 + row][k * 128 + kkk * 32 + kb]);
                        v16bf b = ld_frag(&sWc[col][kkk * 32 + kb]);
                        d = __builtin_amdgcn_wmma_f32_16x16x32_bf16(
                            false, a, false, b, (short)0, d, false, false);
                    }
#pragma unroll
                    for (int r = 0; r < 8; ++r) {
                        int m    = t * 16 + r + mhi;
                        int gm   = base + m;
                        int node = (gm < count) ? s_idx[gm] : 0;
                        float fg = sigmoidf_(d[r] + bf_l[k]);
                        float cv = child_c[node * (Kc * Hd) + k * Hd + col];
                        accFC[t][r] += sMask[m][k] * fg * cv;
                    }
                }
            }
            stage_wait(tid);     // next buffer's DMA done (overlapped w/ compute)
            __syncthreads();
        }

        // ---- Final epilogue ----
#pragma unroll
        for (int t = 0; t < NT; ++t) {
#pragma unroll
            for (int r = 0; r < 8; ++r) {
                int m  = t * 16 + r + mhi;
                int gm = base + m;
                if (gm >= count) continue;
                int node = s_idx[gm];
                float si = accG[0][t][r], so = accG[1][t][r], su = accG[2][t][r];
#pragma unroll
                for (int k = 0; k < Kc; ++k) {
                    float mk = sMask[m][k];
                    si += mk * bi_l[k];
                    so += mk * bo_l[k];
                    su += mk * bu_l[k];
                }
                float iv = sigmoidf_(si);
                float ov = sigmoidf_(so);
                float uv = tanhf(su);
                float cv = iv * uv + accFC[t][r];
                float hv = ov * tanhf(cv);
                out[node * Hd + col]             = hv;   // h
                out[Ntot * Hd + node * Hd + col] = cv;   // c
            }
        }
    }
}

extern "C" void kernel_launch(void* const* d_in, const int* in_sizes, int n_in,
                              void* d_out, int out_size, void* d_ws, size_t ws_size,
                              hipStream_t stream) {
    (void)in_sizes; (void)n_in; (void)out_size; (void)ws_size;  // ws need: 384*128*128*2 B = 12.6 MB
    __bf16* Wt = (__bf16*)d_ws;

    prep_weights<<<4 * Vops * Kc, 256, 0, stream>>>(
        (const float*)d_in[2],   // W_i
        (const float*)d_in[4],   // W_o
        (const float*)d_in[5],   // W_u
        (const float*)d_in[3],   // W_f
        Wt);

    tree_lstm_wmma<<<Vops, 256, 0, stream>>>(
        (const float*)d_in[0],   // child_h
        (const float*)d_in[1],   // child_c
        Wt,
        (const float*)d_in[6],   // b_i
        (const float*)d_in[7],   // b_f
        (const float*)d_in[8],   // b_o
        (const float*)d_in[9],   // b_u
        (const int*)d_in[10],    // ops
        (const int*)d_in[11],    // n_children
        (float*)d_out);
}
